// GraphMatchNet_29764123361609
// MI455X (gfx1250) — compile-verified, matrix operand order checked by reading, not run on hardware
//
#include <hip/hip_runtime.h>
#include <hip/hip_bf16.h>

#define D 256
#define D2 512
#define D3 768
#define NN 8192
#define EE 131072
#define NLAYERS 3

typedef __attribute__((ext_vector_type(16))) __bf16 v16bf;
typedef __attribute__((ext_vector_type(8)))  float  v8f;
typedef __attribute__((ext_vector_type(4)))  int    v4i;

#if defined(__has_builtin)
#  if __has_builtin(__builtin_amdgcn_global_load_async_to_lds_b128)
#    define HAVE_ASYNC_LDS 1
#  endif
#endif
#ifndef HAVE_ASYNC_LDS
#  define HAVE_ASYNC_LDS 0
#endif

__device__ __forceinline__ v8f wmma_bf16(v16bf a, v16bf b, v8f c) {
    return __builtin_amdgcn_wmma_f32_16x16x32_bf16(false, a, false, b, (short)0, c, false, false);
}

union TileU { v16bf v; uint4 q[2]; };

// Load a 16x32 bf16 tile in WMMA A-layout (also used for B tiles sourced from a
// row-major [Dout][K] matrix, i.e. out = X @ W^T).
__device__ __forceinline__ v16bf ld_tile(const __bf16* p, int stride, int row0, int k0, int lane) {
    const __bf16* q = p + (size_t)(row0 + (lane & 15)) * (size_t)stride + k0 + ((lane >> 4) << 3);
    TileU t;
    t.q[0] = *(const uint4*)q;
    t.q[1] = *(const uint4*)(q + 16);
    return t.v;
}

__device__ __forceinline__ void wave_lds_fence() {
    __builtin_amdgcn_wave_barrier();
    asm volatile("s_wait_dscnt 0" ::: "memory");
    __builtin_amdgcn_wave_barrier();
}

__device__ __forceinline__ void async_wait0() {
#if defined(__has_builtin) && __has_builtin(__builtin_amdgcn_s_wait_asynccnt)
    __builtin_amdgcn_s_wait_asynccnt(0);
#else
    asm volatile("s_wait_asynccnt 0" ::: "memory");
#endif
}

#if HAVE_ASYNC_LDS
// Builtin prototype (from compiler diagnostic): param0 is AS1 (global) int4*,
// param1 the LDS-side pointer. AS3 pointers are 32-bit; the low 32 bits of a
// flat shared address are the LDS byte offset per the aperture rules.
__device__ __forceinline__ __attribute__((address_space(1))) v4i* as_g4(const void* p) {
    return (__attribute__((address_space(1))) v4i*)(size_t)p;
}
__device__ __forceinline__ __attribute__((address_space(3))) v4i* as_l4(void* p) {
    return (__attribute__((address_space(3))) v4i*)(unsigned)(size_t)p;
}
__device__ __forceinline__ void async_copy16(const void* gsrc, void* ldst) {
    __builtin_amdgcn_global_load_async_to_lds_b128(as_g4(gsrc), as_l4(ldst), 0, 0);
}
#endif

__device__ __forceinline__ v8f vzero8() {
    v8f z = {0.f, 0.f, 0.f, 0.f, 0.f, 0.f, 0.f, 0.f};
    return z;
}

// ---------------------------------------------------------------- gather/cvt
__global__ void k_cvt_bf16(const float* __restrict__ src, __bf16* __restrict__ dst, int n) {
    int i = blockIdx.x * blockDim.x + threadIdx.x;
    if (i < n) dst[i] = (__bf16)src[i];
}

__global__ void k_zero(float* __restrict__ p, int n) {
    int i = blockIdx.x * blockDim.x + threadIdx.x;
    if (i < n) p[i] = 0.f;
}

__global__ void k_gather(const int* __restrict__ x1, const int* __restrict__ x2,
                         const float* __restrict__ embed,
                         float* __restrict__ h1f, float* __restrict__ h2f,
                         __bf16* __restrict__ h1b, __bf16* __restrict__ h2b,
                         __bf16* __restrict__ h1T, __bf16* __restrict__ h2T) {
    int node = blockIdx.x;
    int g = blockIdx.y;
    const int* x = g ? x2 : x1;
    float* hf = g ? h2f : h1f;
    __bf16* hb = g ? h2b : h1b;
    __bf16* hT = g ? h2T : h1T;
    int row = x[node];
    int d = threadIdx.x;                    // 256 threads = D
    float v = embed[(size_t)row * D + d];
    hf[(size_t)node * D + d] = v;
    __bf16 b = (__bf16)v;
    hb[(size_t)node * D + d] = b;
    hT[(size_t)d * NN + node] = b;
}

// ---------------------------------------------------------------- messages
// Per block: 16 edges. Stage concat rows [h[dst] | h[src] | ew] (16x768 bf16)
// into LDS (async DMA for the two h-row gathers when available), then the
// 768->256 GEMM on WMMA, scatter-add into m[dst] with fp32 atomics.
__global__ void __launch_bounds__(256) k_messages(
        const int* __restrict__ ei1, const int* __restrict__ ei2,
        const int* __restrict__ ea1, const int* __restrict__ ea2,
        const float* __restrict__ edge_embed,
        const __bf16* __restrict__ h1b, const __bf16* __restrict__ h2b,
        const __bf16* __restrict__ Wm_b, const float* __restrict__ bm,
        float* __restrict__ m1, float* __restrict__ m2) {
    __shared__ __align__(16) __bf16 Xs[16 * D3];   // 24 KB
    __shared__ int dsts[16];
    int g = blockIdx.y;
    const int* ei = g ? ei2 : ei1;
    const int* ea = g ? ea2 : ea1;
    const __bf16* hb = g ? h2b : h1b;
    float* m = g ? m2 : m1;
    int e0 = blockIdx.x * 16;
    int tid = threadIdx.x;

#if HAVE_ASYNC_LDS
    // 16 edges x 2 rows x 512B = 1024 16-byte chunks, 4 per thread.
#pragma unroll
    for (int j = 0; j < 4; ++j) {
        int cid = tid + j * 256;
        int e = cid >> 6;                 // edge within tile
        int part = (cid >> 5) & 1;        // 0 = h[dst], 1 = h[src]
        int c16 = cid & 31;               // 16B chunk within 512B row
        int ge = e0 + e;
        int node = part ? ei[ge] : ei[EE + ge];
        async_copy16(hb + (size_t)node * D + c16 * 8,
                     Xs + e * D3 + part * D + c16 * 8);
    }
    // edge embeddings need fp32->bf16 conversion: synchronous path.
    for (int idx = tid; idx < 16 * D; idx += 256) {
        int e = idx >> 8;
        int c = idx & 255;
        int a = ea[e0 + e];
        Xs[e * D3 + 2 * D + c] = (__bf16)edge_embed[(size_t)a * D + c];
    }
    if (tid < 16) dsts[tid] = ei[EE + e0 + tid];
    async_wait0();
    __syncthreads();
#else
    for (int idx = tid; idx < 16 * D3; idx += 256) {
        int e = idx / D3;
        int c = idx - e * D3;
        int ge = e0 + e;
        __bf16 val;
        if (c < D) {
            int dst = ei[EE + ge];                          // edge_index[1]
            val = hb[(size_t)dst * D + c];
        } else if (c < 2 * D) {
            int src = ei[ge];                               // edge_index[0]
            val = hb[(size_t)src * D + (c - D)];
        } else {
            int a = ea[ge];
            val = (__bf16)edge_embed[(size_t)a * D + (c - 2 * D)];
        }
        Xs[idx] = val;
    }
    if (tid < 16) dsts[tid] = ei[EE + e0 + tid];
    __syncthreads();
#endif

    int wave = tid >> 5, lane = tid & 31;
    int n0 = wave * 32;
    v8f acc0 = vzero8(), acc1 = vzero8();
    for (int k = 0; k < D3; k += 32) {
        v16bf a  = ld_tile(Xs, D3, 0, k, lane);
        v16bf b0 = ld_tile(Wm_b, D3, n0,      k, lane);
        v16bf b1 = ld_tile(Wm_b, D3, n0 + 16, k, lane);
        acc0 = wmma_bf16(a, b0, acc0);
        acc1 = wmma_bf16(a, b1, acc1);
    }
    int rb = (lane >> 4) << 3;
    int col0 = n0 + (lane & 15);
    float bias0 = bm[col0], bias1 = bm[col0 + 16];
#pragma unroll
    for (int r = 0; r < 8; ++r) {
        int dst = dsts[rb + r];
        atomicAdd(&m[(size_t)dst * D + col0],      acc0[r] + bias0);
        atomicAdd(&m[(size_t)dst * D + col0 + 16], acc1[r] + bias1);
    }
}

// ---------------------------------------------------------------- attention
__global__ void __launch_bounds__(128) k_attn(
        const __bf16* __restrict__ h1b, const __bf16* __restrict__ h2b,
        const __bf16* __restrict__ h1T, const __bf16* __restrict__ h2T,
        const float* __restrict__ h1f, const float* __restrict__ h2f,
        float* __restrict__ mu1, float* __restrict__ mu2) {
    __shared__ __align__(16) __bf16 Qs[4][16 * D];   // 32 KB
    __shared__ __align__(16) __bf16 Ps[4][16 * 32];  // 4 KB
    int dir = blockIdx.y;
    const __bf16* Qb = dir ? h2b : h1b;
    const __bf16* Kb = dir ? h1b : h2b;
    const __bf16* VT = dir ? h1T : h2T;
    const float*  Qf = dir ? h2f : h1f;
    float* mu = dir ? mu2 : mu1;
    int wave = threadIdx.x >> 5, lane = threadIdx.x & 31;
    int q0 = (blockIdx.x * 4 + wave) * 16;

    {
        const uint4* src = (const uint4*)(Qb + (size_t)q0 * D);
        uint4* dst = (uint4*)Qs[wave];
#if HAVE_ASYNC_LDS
        for (int i = lane; i < 512; i += 32)
            async_copy16(src + i, dst + i);
        async_wait0();
#else
        for (int i = lane; i < 512; i += 32) dst[i] = src[i];
#endif
    }
    wave_lds_fence();

    float mrow[8], lrow[8];
#pragma unroll
    for (int r = 0; r < 8; ++r) { mrow[r] = -3.4e38f; lrow[r] = 0.f; }
    v8f O[16];
#pragma unroll
    for (int t = 0; t < 16; ++t) O[t] = vzero8();

    for (int key0 = 0; key0 < NN; key0 += 32) {
        v8f S0 = vzero8(), S1 = vzero8();
        for (int k = 0; k < D; k += 32) {
            v16bf a  = ld_tile(Qs[wave], D, 0, k, lane);
            v16bf b0 = ld_tile(Kb, D, key0,      k, lane);
            v16bf b1 = ld_tile(Kb, D, key0 + 16, k, lane);
            S0 = wmma_bf16(a, b0, S0);
            S1 = wmma_bf16(a, b1, S1);
        }
        float corr[8];
#pragma unroll
        for (int r = 0; r < 8; ++r) {
            float s0 = S0[r], s1 = S1[r];
            float mx = fmaxf(s0, s1);
#pragma unroll
            for (int off = 1; off < 16; off <<= 1)
                mx = fmaxf(mx, __shfl_xor(mx, off, 32));
            float mnew = fmaxf(mrow[r], mx);
            float p0 = __expf(s0 - mnew);
            float p1 = __expf(s1 - mnew);
            float psum = p0 + p1;
#pragma unroll
            for (int off = 1; off < 16; off <<= 1)
                psum += __shfl_xor(psum, off, 32);
            float c = __expf(mrow[r] - mnew);
            lrow[r] = lrow[r] * c + psum;
            mrow[r] = mnew;
            corr[r] = c;
            S0[r] = p0; S1[r] = p1;
        }
#pragma unroll
        for (int t = 0; t < 16; ++t)
#pragma unroll
            for (int r = 0; r < 8; ++r) O[t][r] = O[t][r] * corr[r];

        wave_lds_fence();
        {
            __bf16* P = Ps[wave];
            int col = lane & 15;
            int rbp = (lane >> 4) << 3;
#pragma unroll
            for (int r = 0; r < 8; ++r) {
                P[(rbp + r) * 32 + col]      = (__bf16)S0[r];
                P[(rbp + r) * 32 + col + 16] = (__bf16)S1[r];
            }
        }
        wave_lds_fence();
        v16bf pa = ld_tile(Ps[wave], 32, 0, 0, lane);
#pragma unroll
        for (int t = 0; t < 16; ++t) {
            v16bf b = ld_tile(VT, NN, t * 16, key0, lane);
            O[t] = wmma_bf16(pa, b, O[t]);
        }
    }

    float inv[8];
#pragma unroll
    for (int r = 0; r < 8; ++r) inv[r] = 1.0f / lrow[r];
    int rb = (lane >> 4) << 3;
#pragma unroll
    for (int t = 0; t < 16; ++t) {
        int col = t * 16 + (lane & 15);
#pragma unroll
        for (int r = 0; r < 8; ++r) {
            int row = q0 + rb + r;
            mu[(size_t)row * D + col] =
                Qf[(size_t)row * D + col] - O[t][r] * inv[r];
        }
    }
}

// ---------------------------------------------------------------- fused GRU
__global__ void __launch_bounds__(256) k_gru(
        const float* __restrict__ m1, const float* __restrict__ m2,
        const float* __restrict__ mu1, const float* __restrict__ mu2,
        const float* __restrict__ h1f, const float* __restrict__ h2f,
        const __bf16* __restrict__ h1b, const __bf16* __restrict__ h2b,
        const __bf16* __restrict__ Wih_b, const __bf16* __restrict__ Whh_b,
        const float* __restrict__ b_ih, const float* __restrict__ b_hh,
        float* __restrict__ h1nf, float* __restrict__ h2nf,
        __bf16* __restrict__ h1nb, __bf16* __restrict__ h2nb,
        __bf16* __restrict__ h1nT, __bf16* __restrict__ h2nT) {
    __shared__ __align__(16) __bf16 Xs[16 * D2];   // 16 KB
    int g = blockIdx.y;
    const float* m  = g ? m2 : m1;
    const float* mu = g ? mu2 : mu1;
    const float* hf = g ? h2f : h1f;
    const __bf16* hb = g ? h2b : h1b;
    float* hnf = g ? h2nf : h1nf;
    __bf16* hnb = g ? h2nb : h1nb;
    __bf16* hnT = g ? h2nT : h1nT;
    int n0 = blockIdx.x * 16;
    int tid = threadIdx.x;

    for (int i = tid; i < 16 * D2; i += 256) {
        int row = i >> 9;
        int c = i & (D2 - 1);
        float v = (c < D) ? m[(size_t)(n0 + row) * D + c]
                          : mu[(size_t)(n0 + row) * D + (c - D)];
        Xs[i] = (__bf16)v;
    }
    __syncthreads();

    int wave = tid >> 5, lane = tid & 31;
    int c0 = wave * 32;
    v8f aR[2], aZ[2], aNi[2], aNh[2];
#pragma unroll
    for (int t = 0; t < 2; ++t) { aR[t] = vzero8(); aZ[t] = vzero8(); aNi[t] = vzero8(); aNh[t] = vzero8(); }

    for (int k = 0; k < D2; k += 32) {                 // gi = [m|mu] @ W_ih^T
        v16bf a = ld_tile(Xs, D2, 0, k, lane);
#pragma unroll
        for (int t = 0; t < 2; ++t) {
            int cc = c0 + t * 16;
            aR[t]  = wmma_bf16(a, ld_tile(Wih_b, D2, cc,         k, lane), aR[t]);
            aZ[t]  = wmma_bf16(a, ld_tile(Wih_b, D2, cc + D,     k, lane), aZ[t]);
            aNi[t] = wmma_bf16(a, ld_tile(Wih_b, D2, cc + 2 * D, k, lane), aNi[t]);
        }
    }
    for (int k = 0; k < D; k += 32) {                  // gh = h @ W_hh^T
        v16bf a = ld_tile(hb, D, n0, k, lane);
#pragma unroll
        for (int t = 0; t < 2; ++t) {
            int cc = c0 + t * 16;
            aR[t]  = wmma_bf16(a, ld_tile(Whh_b, D, cc,         k, lane), aR[t]);
            aZ[t]  = wmma_bf16(a, ld_tile(Whh_b, D, cc + D,     k, lane), aZ[t]);
            aNh[t] = wmma_bf16(a, ld_tile(Whh_b, D, cc + 2 * D, k, lane), aNh[t]);
        }
    }

    int rb = (lane >> 4) << 3;
#pragma unroll
    for (int t = 0; t < 2; ++t) {
        int col = c0 + t * 16 + (lane & 15);
        float br = b_ih[col]     + b_hh[col];
        float bz = b_ih[col + D] + b_hh[col + D];
        float bin = b_ih[col + 2 * D];
        float bhn = b_hh[col + 2 * D];
#pragma unroll
        for (int r8 = 0; r8 < 8; ++r8) {
            int row = n0 + rb + r8;
            float rg = 1.f / (1.f + __expf(-(aR[t][r8] + br)));
            float zg = 1.f / (1.f + __expf(-(aZ[t][r8] + bz)));
            float ng = tanhf(aNi[t][r8] + bin + rg * (aNh[t][r8] + bhn));
            float hprev = hf[(size_t)row * D + col];
            float hnew = (1.f - zg) * ng + zg * hprev;
            hnf[(size_t)row * D + col] = hnew;
            __bf16 b16 = (__bf16)hnew;
            hnb[(size_t)row * D + col] = b16;
            hnT[(size_t)col * NN + row] = b16;
        }
    }
}

// ---------------------------------------------------------------- pooling
__global__ void __launch_bounds__(256) k_pool(
        const float* __restrict__ h1f, const float* __restrict__ h2f,
        const float* __restrict__ Wg, const float* __restrict__ bg,
        float* __restrict__ out) {
    __shared__ float logits[NN];   // 32 KB
    __shared__ float red[256];
    int g = blockIdx.x;
    const float* h = g ? h2f : h1f;
    int tid = threadIdx.x;
    float bgv = bg[0];
    for (int n = tid; n < NN; n += 256) {
        float dot = 0.f;
        const float* row = h + (size_t)n * D;
        for (int d = 0; d < D; ++d) dot += row[d] * Wg[d];
        logits[n] = 1.f / (1.f + __expf(-(dot + bgv)));
    }
    __syncthreads();
    float mx = -3.4e38f;
    for (int n = tid; n < NN; n += 256) mx = fmaxf(mx, logits[n]);
    red[tid] = mx; __syncthreads();
    for (int s = 128; s > 0; s >>= 1) {
        if (tid < s) red[tid] = fmaxf(red[tid], red[tid + s]);
        __syncthreads();
    }
    mx = red[0]; __syncthreads();
    float sm = 0.f;
    for (int n = tid; n < NN; n += 256) {
        float e = __expf(logits[n] - mx);
        logits[n] = e;
        sm += e;
    }
    __syncthreads();
    red[tid] = sm; __syncthreads();
    for (int s = 128; s > 0; s >>= 1) {
        if (tid < s) red[tid] += red[tid + s];
        __syncthreads();
    }
    float inv = 1.f / red[0];
    __syncthreads();
    float acc = 0.f;
    for (int n = 0; n < NN; ++n) acc += logits[n] * h[(size_t)n * D + tid];
    out[g * D + tid] = acc * inv;
}

// ---------------------------------------------------------------- launcher
extern "C" void kernel_launch(void* const* d_in, const int* in_sizes, int n_in,
                              void* d_out, int out_size, void* d_ws, size_t ws_size,
                              hipStream_t stream) {
    const int* x1  = (const int*)d_in[0];
    const int* x2  = (const int*)d_in[1];
    const int* ei1 = (const int*)d_in[2];
    const int* ei2 = (const int*)d_in[3];
    const int* ea1 = (const int*)d_in[4];
    const int* ea2 = (const int*)d_in[5];
    const float* embed      = (const float*)d_in[6];
    const float* edge_embed = (const float*)d_in[7];
    const float* Wm   = (const float*)d_in[8];
    const float* bm   = (const float*)d_in[9];
    const float* W_ih = (const float*)d_in[10];
    const float* W_hh = (const float*)d_in[11];
    const float* b_ih = (const float*)d_in[12];
    const float* b_hh = (const float*)d_in[13];
    const float* Wg   = (const float*)d_in[14];
    const float* bg   = (const float*)d_in[15];
    float* out = (float*)d_out;

    char* w = (char*)d_ws;
    size_t off = 0;
    auto alloc = [&](size_t bytes) -> void* {
        void* p = w + off;
        off = (off + bytes + 255) & ~(size_t)255;
        return p;
    };

    float *h1f[2], *h2f[2];
    __bf16 *h1b[2], *h2b[2], *h1T[2], *h2T[2];
    for (int i = 0; i < 2; ++i) {
        h1f[i] = (float*)alloc((size_t)NN * D * 4);
        h2f[i] = (float*)alloc((size_t)NN * D * 4);
        h1b[i] = (__bf16*)alloc((size_t)NN * D * 2);
        h2b[i] = (__bf16*)alloc((size_t)NN * D * 2);
        h1T[i] = (__bf16*)alloc((size_t)NN * D * 2);
        h2T[i] = (__bf16*)alloc((size_t)NN * D * 2);
    }
    float* msg = (float*)alloc((size_t)2 * NN * D * 4);
    float* m1 = msg;
    float* m2 = msg + (size_t)NN * D;
    float* mu1 = (float*)alloc((size_t)NN * D * 4);
    float* mu2 = (float*)alloc((size_t)NN * D * 4);
    __bf16* Wm_b  = (__bf16*)alloc((size_t)D * D3 * 2);
    __bf16* Wih_b = (__bf16*)alloc((size_t)D3 * D2 * 2);
    __bf16* Whh_b = (__bf16*)alloc((size_t)D3 * D * 2);

    k_cvt_bf16<<<(D * D3 + 255) / 256, 256, 0, stream>>>(Wm, Wm_b, D * D3);
    k_cvt_bf16<<<(D3 * D2 + 255) / 256, 256, 0, stream>>>(W_ih, Wih_b, D3 * D2);
    k_cvt_bf16<<<(D3 * D + 255) / 256, 256, 0, stream>>>(W_hh, Whh_b, D3 * D);
    k_gather<<<dim3(NN, 2), 256, 0, stream>>>(x1, x2, embed,
        h1f[0], h2f[0], h1b[0], h2b[0], h1T[0], h2T[0]);

    int cur = 0;
    for (int layer = 0; layer < NLAYERS; ++layer) {
        int nxt = cur ^ 1;
        k_zero<<<(2 * NN * D + 255) / 256, 256, 0, stream>>>(msg, 2 * NN * D);
        k_messages<<<dim3(EE / 16, 2), 256, 0, stream>>>(
            ei1, ei2, ea1, ea2, edge_embed,
            h1b[cur], h2b[cur], Wm_b, bm, m1, m2);
        k_attn<<<dim3(NN / 64, 2), 128, 0, stream>>>(
            h1b[cur], h2b[cur], h1T[cur], h2T[cur],
            h1f[cur], h2f[cur], mu1, mu2);
        k_gru<<<dim3(NN / 16, 2), 256, 0, stream>>>(
            m1, m2, mu1, mu2,
            h1f[cur], h2f[cur], h1b[cur], h2b[cur],
            Wih_b, Whh_b, b_ih, b_hh,
            h1f[nxt], h2f[nxt], h1b[nxt], h2b[nxt],
            h1T[nxt], h2T[nxt]);
        cur = nxt;
    }
    k_pool<<<2, 256, 0, stream>>>(h1f[cur], h2f[cur], Wg, bg, out);
}
